// GCN_49452253446476
// MI455X (gfx1250) — matrix operand order repaired and checked by reading, not compiled, not run on hardware
//
#include <hip/hip_runtime.h>
#include <hip/hip_bf16.h>

typedef __attribute__((ext_vector_type(16))) _Float16 v16h;
typedef __attribute__((ext_vector_type(8)))  float    v8f;

#define NNODES 10000
#define NEDGES 320000
#define TILE   16
#define AS     328   // conv A-tile row stride in halves (320 + 8 pad, kills LDS bank conflicts)

// ---------------------------------------------------------------------------
// WMMA fragment builders (CDNA5 16x16x32 f16 layouts, CDNA5 ISA)
// A 16x32 f16: lane L -> row M=L&15; VGPR p holds K 2p,2p+1; upper lane half +16,
// (contiguous K, same order as B).  B 32x16 f16: lane L -> col n=L&15, lanes>=16 hold
// K=16..31; 16 contiguous K halves per lane -> one 32B v16h load.
// C/D f32: lane L, VGPR v -> M = v + 8*(L>=16), N = L&15.
// ---------------------------------------------------------------------------
__device__ __forceinline__ v16h a_frag_row(const _Float16* __restrict__ row, int lane) {
  const int hi = (lane & 16) ? 16 : 0;
  v16h a;
#pragma unroll
  for (int p = 0; p < 8; ++p) {
    const int kb = 2 * p + hi;   // 2*p + {0|16}
    a[2 * p]     = row[kb];
    a[2 * p + 1] = row[kb + 1];
  }
  return a;
}

__device__ __forceinline__ v16h a_frag_row_f32(const float* __restrict__ row, float scale, int lane) {
  const int hi = (lane & 16) ? 16 : 0;
  v16h a;
#pragma unroll
  for (int p = 0; p < 8; ++p) {
    const int kb = 2 * p + hi;
    a[2 * p]     = (_Float16)(row[kb] * scale);
    a[2 * p + 1] = (_Float16)(row[kb + 1] * scale);
  }
  return a;
}

// ---------------------------------------------------------------------------
// Prep: transpose + f32->f16 convert all GEMM weights so B fragments are
// contiguous 32B loads.  w1t[n][K], w2t[n][k], s1st/s1nt[n][k], s2st/s2nt[n][k]
// (s2 padded from 10 to 16 output columns with zeros).
// ---------------------------------------------------------------------------
__global__ __launch_bounds__(256)
void prep_kernel(const float* __restrict__ w1, const float* __restrict__ w2,
                 const float* __restrict__ s1s, const float* __restrict__ s1n,
                 const float* __restrict__ s2s, const float* __restrict__ s2n,
                 _Float16* __restrict__ w1t, _Float16* __restrict__ w2t,
                 _Float16* __restrict__ s1st, _Float16* __restrict__ s1nt,
                 _Float16* __restrict__ s2st, _Float16* __restrict__ s2nt) {
  const int i  = blockIdx.x * 256 + threadIdx.x;
  const int gs = gridDim.x * 256;
  for (int idx = i; idx < 32 * 10240; idx += gs) {         // FC1: [10240,32] -> [32][10240]
    const int n = idx / 10240, K = idx - n * 10240;
    w1t[idx] = (_Float16)w1[(size_t)K * 32 + n];
  }
  if (i < 1024) {                                          // FC2: [32,32] -> [32][32]
    const int n = i >> 5, k = i & 31;
    w2t[i] = (_Float16)w2[k * 32 + n];
  }
  if (i < 2048) {                                          // SAGE1: [32,64] -> [64][32]
    const int n = i >> 5, k = i & 31;
    s1st[i] = (_Float16)s1s[k * 64 + n];
    s1nt[i] = (_Float16)s1n[k * 64 + n];
  }
  if (i < 1024) {                                          // SAGE2: [64,10] -> [16][64] padded
    const int n = i >> 6, k = i & 63;
    s2st[i] = (n < 10) ? (_Float16)s2s[k * 10 + n] : (_Float16)0.0f;
    s2nt[i] = (n < 10) ? (_Float16)s2n[k * 10 + n] : (_Float16)0.0f;
  }
}

__global__ __launch_bounds__(256)
void zero_kernel(float* __restrict__ p, int n) {
  const int i  = blockIdx.x * 256 + threadIdx.x;
  const int gs = gridDim.x * 256;
  for (int idx = i; idx < n; idx += gs) p[idx] = 0.0f;
}

// ---------------------------------------------------------------------------
// Fused CNN: conv3x3(1->32)+ReLU+maxpool2x2 per output channel -> 16x320 f16
// A-tile in LDS -> WMMA into FC1 accumulators (K=10240 over 32 channels x 10
// chunks) -> bias+ReLU -> FC2 via WMMA -> bias+ReLU -> h16[N,32] f16.
// 16 nodes per block, 256 threads (8 wave32).  Wave IDs are scalarized with
// readfirstlane so every WMMA executes under uniform SALU control flow
// (EXEC all-ones, per CDNA5 ISA restrictions).
// ---------------------------------------------------------------------------
__global__ __launch_bounds__(256)
void cnn_fc_kernel(const float* __restrict__ feat, const float* __restrict__ convw,
                   const float* __restrict__ convb, const _Float16* __restrict__ w1t,
                   const float* __restrict__ b1, const _Float16* __restrict__ w2t,
                   const float* __restrict__ b2, _Float16* __restrict__ h16) {
  __shared__ _Float16 sF[TILE * 1280];   // 40 KB: 16 nodes x 20x64 features, f16
  __shared__ _Float16 sA[TILE * AS];     // 10.25 KB: per-channel pooled A-tile (reused for h1)
  __shared__ float    sW[288];           // conv weights [32][9]
  __shared__ float    sCB[32];           // conv bias
  __shared__ float    sRed[8 * 32 * 8];  // 8 KB: partial accumulator reduction

  const int tid   = threadIdx.x;
  const int wave  = __builtin_amdgcn_readfirstlane(tid >> 5);  // scalar wave id
  const int lane  = tid & 31;
  const int node0 = blockIdx.x * TILE;

  if (tid < 288) sW[tid]  = convw[tid];
  if (tid < 32)  sCB[tid] = convb[tid];

  // Stage features (f32 global -> f16 LDS), 5120 float4 per tile.
  const float* gf = feat + (size_t)node0 * 1280;
#pragma unroll
  for (int i = 0; i < 20; ++i) {
    const int idx = tid + i * 256;
    const float4 v = ((const float4*)gf)[idx];
    sF[idx * 4 + 0] = (_Float16)v.x;
    sF[idx * 4 + 1] = (_Float16)v.y;
    sF[idx * 4 + 2] = (_Float16)v.z;
    sF[idx * 4 + 3] = (_Float16)v.w;
  }
  __syncthreads();

  v8f acc = {};
  const int cb2 = wave & 1;   // scalar: output column block (cols 0..15 / 16..31)
  const int jg  = wave >> 1;  // scalar: K-chunk group -> chunks jg, jg+4, jg+8
  const int nB  = cb2 * 16 + (lane & 15);          // this lane's B column
  const int khB = (lane & 16) ? 16 : 0;            // B K-half offset
  const _Float16* wrow = w1t + (size_t)nB * 10240 + khB;

  for (int c = 0; c < 32; ++c) {
    // Prefetch this channel's B chunks into cache; the conv phase below gives
    // the prefetch a long lead time (lowers to global_prefetch_b8).
    for (int j = jg; j < 10; j += 4)
      __builtin_prefetch(wrow + c * 320 + j * 32, 0, 0);

    float wr[9];
#pragma unroll
    for (int q = 0; q < 9; ++q) wr[q] = sW[c * 9 + q];
    const float bc = sCB[c];

    // conv + ReLU + 2x2 maxpool for channel c: 16 nodes x 10x32 pooled outputs.
    for (int i = 0; i < 20; ++i) {
      const int idx = tid + i * 256;            // 0..5119
      const int nd  = idx / 320;
      const int rem = idx - nd * 320;
      const int ph  = rem >> 5;
      const int pw  = rem & 31;
      const _Float16* fb = sF + nd * 1280;
      float m = 0.0f;                           // post-ReLU max (ReLU => >= 0)
#pragma unroll
      for (int dy = 0; dy < 2; ++dy) {
#pragma unroll
        for (int dx = 0; dx < 2; ++dx) {
          const int y = 2 * ph + dy, x = 2 * pw + dx;
          float s = bc;
#pragma unroll
          for (int ky = -1; ky <= 1; ++ky) {
            const int yy = y + ky;
            if (yy < 0 || yy >= 20) continue;
#pragma unroll
            for (int kx = -1; kx <= 1; ++kx) {
              const int xx = x + kx;
              if (xx < 0 || xx >= 64) continue;
              s += wr[(ky + 1) * 3 + (kx + 1)] * (float)fb[yy * 64 + xx];
            }
          }
          m = fmaxf(m, s);
        }
      }
      sA[nd * AS + rem] = (_Float16)m;
    }
    __syncthreads();

    // FC1 WMMA: this channel contributes K = c*320 .. c*320+319 (10 chunks of 32).
    // jg is scalar -> uniform SALU loop, EXEC untouched.
    for (int j = jg; j < 10; j += 4) {
      const v16h a = a_frag_row(sA + (lane & 15) * AS + j * 32, lane);
      const v16h b = *(const v16h*)(wrow + c * 320 + j * 32);
      acc = __builtin_amdgcn_wmma_f32_16x16x32_f16(false, a, false, b, (short)0, acc,
                                                   false, false);
    }
    __syncthreads();
  }

  // Reduce the 4 partial accumulators per column block.
#pragma unroll
  for (int v = 0; v < 8; ++v) sRed[(wave * 32 + lane) * 8 + v] = acc[v];
  __syncthreads();

#pragma unroll
  for (int i = 0; i < 2; ++i) {
    const int el  = tid + i * 256;              // 0..511 = (M,n) of FC1 output
    const int n   = el & 31;
    const int M   = el >> 5;
    const int cbk = n >> 4;
    const int ln  = (n & 15) + ((M & 8) ? 16 : 0);
    const int v   = M & 7;
    float s = 0.0f;
#pragma unroll
    for (int g = 0; g < 4; ++g) s += sRed[(((g << 1) | cbk) * 32 + ln) * 8 + v];
    s = fmaxf(s + b1[n], 0.0f);
    sA[M * 40 + n] = (_Float16)s;               // h1 tile, stride 40 halves
  }
  __syncthreads();

  // FC2 (K=32, 32 outputs): waves 0 and 1 each do one WMMA column block.
  // wave is scalar -> s_cbranch, not EXEC masking, around the WMMA.
  if (wave < 2) {
    const v16h a = a_frag_row(sA + (lane & 15) * 40, lane);
    const int n  = wave * 16 + (lane & 15);
    const v16h b = *(const v16h*)(w2t + n * 32 + khB);
    v8f acc2 = {};
    acc2 = __builtin_amdgcn_wmma_f32_16x16x32_f16(false, a, false, b, (short)0, acc2,
                                                  false, false);
#pragma unroll
    for (int v = 0; v < 8; ++v) {
      const int M = v + ((lane & 16) ? 8 : 0);
      const float val = fmaxf(acc2[v] + b2[n], 0.0f);
      h16[(size_t)(node0 + M) * 32 + n] = (_Float16)val;
    }
  }
}

// ---------------------------------------------------------------------------
// Edge aggregation: one lane per (edge, channel); f32 global atomics.
// ---------------------------------------------------------------------------
__global__ __launch_bounds__(256)
void agg1_kernel(const int* __restrict__ src, const int* __restrict__ dst,
                 const _Float16* __restrict__ h16, float* __restrict__ agg1,
                 float* __restrict__ deg) {
  const int idx = blockIdx.x * 256 + threadIdx.x;
  const int e = idx >> 5, c = idx & 31;
  if (e >= NEDGES) return;
  const int s = src[e], d = dst[e];
  atomicAdd(agg1 + (size_t)d * 32 + c, (float)h16[(size_t)s * 32 + c]);
  if (c == 0) atomicAdd(deg + d, 1.0f);
}

__global__ __launch_bounds__(256)
void agg2_kernel(const int* __restrict__ src, const int* __restrict__ dst,
                 const _Float16* __restrict__ h1, float* __restrict__ agg2) {
  const int idx = blockIdx.x * 256 + threadIdx.x;
  const int e = idx >> 6, c = idx & 63;
  if (e >= NEDGES) return;
  const int s = src[e], d = dst[e];
  atomicAdd(agg2 + (size_t)d * 64 + c, (float)h1[(size_t)s * 64 + c]);
}

// ---------------------------------------------------------------------------
// SAGE1: h1 = relu(h @ s1_self + (agg1/deg) @ s1_neigh + b1), 32->64.
// 32 nodes per block: wave -> (node subtile, 16-col block); K=32 -> 1 chunk,
// two WMMAs (self + neighbor) per wave.  Wave id scalarized.
// ---------------------------------------------------------------------------
__global__ __launch_bounds__(256)
void sage1_kernel(const _Float16* __restrict__ h16, const float* __restrict__ agg1,
                  const float* __restrict__ deg, const _Float16* __restrict__ s1st,
                  const _Float16* __restrict__ s1nt, const float* __restrict__ s1b,
                  _Float16* __restrict__ h1) {
  const int tid  = threadIdx.x;
  const int wave = __builtin_amdgcn_readfirstlane(tid >> 5);
  const int lane = tid & 31;
  const int sub = wave >> 2, cb = wave & 3;     // scalar
  const int node0 = blockIdx.x * 32 + sub * 16; // scalar
  const int M = lane & 15;
  int nm = node0 + M;
  if (nm >= NNODES) nm = NNODES - 1;            // clamp loads; stores guarded below

  const v16h as = a_frag_row(h16 + (size_t)nm * 32, lane);
  const float rd = 1.0f / fmaxf(deg[nm], 1.0f);
  const v16h an = a_frag_row_f32(agg1 + (size_t)nm * 32, rd, lane);

  const int n  = cb * 16 + M;
  const int kh = (lane & 16) ? 16 : 0;
  const v16h bs = *(const v16h*)(s1st + n * 32 + kh);
  const v16h bn = *(const v16h*)(s1nt + n * 32 + kh);

  v8f acc = {};
  acc = __builtin_amdgcn_wmma_f32_16x16x32_f16(false, as, false, bs, (short)0, acc,
                                               false, false);
  acc = __builtin_amdgcn_wmma_f32_16x16x32_f16(false, an, false, bn, (short)0, acc,
                                               false, false);
  const float bias = s1b[n];
#pragma unroll
  for (int v = 0; v < 8; ++v) {
    const int Mo = v + ((lane & 16) ? 8 : 0);
    const int node = node0 + Mo;
    if (node < NNODES)
      h1[(size_t)node * 64 + n] = (_Float16)fmaxf(acc[v] + bias, 0.0f);
  }
}

// ---------------------------------------------------------------------------
// SAGE2: out = h1 @ s2_self + (agg2/deg) @ s2_neigh + b2, 64->10 (padded 16).
// 16 nodes per block, 2 waves: wave0 = self term, wave1 = neighbor term;
// K=64 -> 2 chunks each; partials reduced through LDS.  The wave split is a
// scalar branch (readfirstlane) so neither side's WMMAs issue with EXEC==0.
// ---------------------------------------------------------------------------
__global__ __launch_bounds__(64)
void sage2_kernel(const _Float16* __restrict__ h1, const float* __restrict__ agg2,
                  const float* __restrict__ deg, const _Float16* __restrict__ s2st,
                  const _Float16* __restrict__ s2nt, const float* __restrict__ s2b,
                  float* __restrict__ out) {
  __shared__ float sRed[2 * 32 * 8];
  const int tid  = threadIdx.x;
  const int wave = __builtin_amdgcn_readfirstlane(tid >> 5);
  const int lane = tid & 31;
  const int node0 = blockIdx.x * 16;
  const int node = node0 + (lane & 15);        // grid exact: 625*16 == N
  const int kh = (lane & 16) ? 16 : 0;
  const int nc = lane & 15;

  v8f acc = {};
  if (wave == 0) {
    const _Float16* row = h1 + (size_t)node * 64;
#pragma unroll
    for (int chunk = 0; chunk < 64; chunk += 32) {
      const v16h a = a_frag_row(row + chunk, lane);
      const v16h b = *(const v16h*)(s2st + nc * 64 + chunk + kh);
      acc = __builtin_amdgcn_wmma_f32_16x16x32_f16(false, a, false, b, (short)0, acc,
                                                   false, false);
    }
  } else {
    const float rd = 1.0f / fmaxf(deg[node], 1.0f);
    const float* row = agg2 + (size_t)node * 64;
#pragma unroll
    for (int chunk = 0; chunk < 64; chunk += 32) {
      const v16h a = a_frag_row_f32(row + chunk, rd, lane);
      const v16h b = *(const v16h*)(s2nt + nc * 64 + chunk + kh);
      acc = __builtin_amdgcn_wmma_f32_16x16x32_f16(false, a, false, b, (short)0, acc,
                                                   false, false);
    }
  }
#pragma unroll
  for (int v = 0; v < 8; ++v) sRed[(wave * 32 + lane) * 8 + v] = acc[v];
  __syncthreads();

  if (wave == 0) {
#pragma unroll
    for (int v = 0; v < 8; ++v) {
      const float s = sRed[lane * 8 + v] + sRed[256 + lane * 8 + v];
      const int Mo = v + ((lane & 16) ? 8 : 0);
      if (nc < 10) out[(size_t)(node0 + Mo) * 10 + nc] = s + s2b[nc];
    }
  }
}

// ---------------------------------------------------------------------------
// Launch: prep -> zero -> CNN -> agg1/deg -> SAGE1 -> agg2 -> SAGE2
// ---------------------------------------------------------------------------
extern "C" void kernel_launch(void* const* d_in, const int* in_sizes, int n_in,
                              void* d_out, int out_size, void* d_ws, size_t ws_size,
                              hipStream_t stream) {
  const float* feat   = (const float*)d_in[0];
  const int*   src    = (const int*)d_in[1];
  const int*   dst    = (const int*)d_in[2];
  const float* conv_w = (const float*)d_in[3];
  const float* conv_b = (const float*)d_in[4];
  const float* w1     = (const float*)d_in[5];
  const float* b1     = (const float*)d_in[6];
  const float* w2     = (const float*)d_in[7];
  const float* b2     = (const float*)d_in[8];
  const float* s1s    = (const float*)d_in[9];
  const float* s1n    = (const float*)d_in[10];
  const float* s1b    = (const float*)d_in[11];
  const float* s2s    = (const float*)d_in[12];
  const float* s2n    = (const float*)d_in[13];
  const float* s2b    = (const float*)d_in[14];
  float* out = (float*)d_out;

  // Workspace layout (all region offsets multiples of 256B).
  char* ws = (char*)d_ws;
  _Float16* w1t  = (_Float16*)(ws);                 // 32 x 10240 f16   = 655360 B
  _Float16* w2t  = (_Float16*)(ws + 655360);        // 32 x 32 f16      =   2048 B
  _Float16* s1st = (_Float16*)(ws + 657408);        // 64 x 32 f16      =   4096 B
  _Float16* s1nt = (_Float16*)(ws + 661504);        // 64 x 32 f16      =   4096 B
  _Float16* s2st = (_Float16*)(ws + 665600);        // 16 x 64 f16      =   2048 B
  _Float16* s2nt = (_Float16*)(ws + 667648);        // 16 x 64 f16      =   2048 B
  _Float16* h16  = (_Float16*)(ws + 669696);        // N x 32 f16       = 640000 B
  _Float16* h1   = (_Float16*)(ws + 1309696);       // N x 64 f16       = 1280000 B
  float*    agg1 = (float*)(ws + 2589696);          // N x 32 f32       = 1280000 B
  float*    agg2 = (float*)(ws + 3869696);          // N x 64 f32       = 2560000 B
  float*    deg  = (float*)(ws + 6429696);          // N f32            =   40000 B
  // total 6,469,696 B

  prep_kernel<<<1280, 256, 0, stream>>>(w1, w2, s1s, s1n, s2s, s2n,
                                        w1t, w2t, s1st, s1nt, s2st, s2nt);

  // agg1, agg2, deg are contiguous: zero 970000 floats in one pass.
  zero_kernel<<<3790, 256, 0, stream>>>(agg1, 970000);

  cnn_fc_kernel<<<NNODES / TILE, 256, 0, stream>>>(feat, conv_w, conv_b, w1t, b1,
                                                   w2t, b2, h16);

  agg1_kernel<<<(NEDGES * 32) / 256, 256, 0, stream>>>(src, dst, h16, agg1, deg);

  sage1_kernel<<<(NNODES + 31) / 32, 256, 0, stream>>>(h16, agg1, deg, s1st, s1nt,
                                                       s1b, h1);

  agg2_kernel<<<(NEDGES * 64) / 256, 256, 0, stream>>>(src, dst, h1, agg2);

  sage2_kernel<<<NNODES / 16, 64, 0, stream>>>(h1, agg2, deg, s2st, s2nt, s2b, out);
}
